// CRFModel_27513560498640
// MI455X (gfx1250) — matrix operand order repaired and checked by reading, not compiled, not run on hardware
//
#include <hip/hip_runtime.h>
#include <hip/hip_bf16.h>
#include <math.h>

typedef __attribute__((ext_vector_type(16))) _Float16 v16h;
typedef __attribute__((ext_vector_type(8)))  float    v8f;

#define NC 16
#define BB 2
#define HH 512
#define WW 512
#define HL 128
#define WL 128
#define KS 11
#define K2 121
#define PADR 5
#define UNARY_W 0.8f
#define PW0 2.0f
#define PW1 0.6f
#define HWF (HH*WW)      // 262144
#define HWL (HL*WL)      // 16384

// tile geometry for the message kernel: 16 px wide x 8 rows per block
#define TROWS 18         // 8 + 2*5 halo rows
#define TCOLS 26         // 16 + 2*5 halo cols
#define TPIX  (TROWS*TCOLS)          // 468
#define TILE_FLOATS (4*TROWS*TCOLS*4) // [quarter][row][col][4ch] = 7488

#if defined(__has_builtin)
# if __has_builtin(__builtin_amdgcn_global_load_async_to_lds_b128)
#  define HAVE_ASYNC_LDS 1
# endif
# if __has_builtin(__builtin_amdgcn_s_wait_asynccnt)
#  define HAVE_WAIT_ASYNC 1
# endif
#endif

// pointer types required by the async builtin (per hipcc diagnostic: the
// global operand is `int __attribute__((vector_size(16))) addrspace(1)*`)
typedef int a_v4i __attribute__((vector_size(16)));
typedef __attribute__((address_space(1))) a_v4i* g_b128_ptr;
typedef __attribute__((address_space(3))) a_v4i* l_b128_ptr;

__device__ __forceinline__ void wait_asynccnt0() {
#if defined(HAVE_WAIT_ASYNC)
    __builtin_amdgcn_s_wait_asynccnt(0);
#elif defined(HAVE_ASYNC_LDS)
    asm volatile("s_wait_asynccnt 0x0" ::: "memory");
#endif
}

// ---------------------------------------------------------------------------
// 1) pool image -> imgLow (B,3,128,128), already divided by RGB_SCALE=13
__global__ void k_pool_image(const float* __restrict__ img, float* __restrict__ imgLow) {
    int idx = blockIdx.x * blockDim.x + threadIdx.x;
    if (idx >= BB * 3 * HWL) return;
    int xl = idx & (WL - 1);
    int yl = (idx >> 7) & (HL - 1);
    int ch = (idx / HWL) % 3;
    int b  = idx / (3 * HWL);
    const float* src = img + ((size_t)(b * 3 + ch)) * HWF;
    float s = 0.f;
    #pragma unroll
    for (int sy = 0; sy < 4; ++sy)
        #pragma unroll
        for (int sx = 0; sx < 4; ++sx)
            s += src[(4 * yl + sy) * WW + (4 * xl + sx)];
    imgLow[idx] = s * (1.f / 16.f) * (1.f / 13.f);
}

// ---------------------------------------------------------------------------
// 2) pixel-adaptive Gaussian kernels K0 (YXRGB/80,13) and K1 (pos/3),
//    layout (B, K2, HL, WL), zero-padded patch features outside the image.
__global__ void k_gauss(const float* __restrict__ imgLow,
                        float* __restrict__ K0, float* __restrict__ K1) {
    int idx = blockIdx.x * blockDim.x + threadIdx.x;
    if (idx >= BB * HWL) return;
    int xl = idx & (WL - 1);
    int yl = (idx >> 7) & (HL - 1);
    int b  = idx >> 14;
    const float* iL = imgLow + (size_t)b * 3 * HWL;

    float fy0 = (4.f * yl + 1.5f) * (1.f / 80.f);
    float fx0 = (4.f * xl + 1.5f) * (1.f / 80.f);
    float r0  = iL[0 * HWL + yl * WL + xl];
    float g0  = iL[1 * HWL + yl * WL + xl];
    float b0  = iL[2 * HWL + yl * WL + xl];
    float fy1 = (4.f * yl + 1.5f) * (1.f / 3.f);
    float fx1 = (4.f * xl + 1.5f) * (1.f / 3.f);

    for (int k = 0; k < K2; ++k) {
        int dy = k / KS - PADR, dx = k % KS - PADR;
        int yn = yl + dy, xn = xl + dx;
        float gy0 = 0.f, gx0 = 0.f, gr = 0.f, gg = 0.f, gb = 0.f;
        float gy1 = 0.f, gx1 = 0.f;
        if (yn >= 0 && yn < HL && xn >= 0 && xn < WL) {
            gy0 = (4.f * yn + 1.5f) * (1.f / 80.f);
            gx0 = (4.f * xn + 1.5f) * (1.f / 80.f);
            gr  = iL[0 * HWL + yn * WL + xn];
            gg  = iL[1 * HWL + yn * WL + xn];
            gb  = iL[2 * HWL + yn * WL + xn];
            gy1 = (4.f * yn + 1.5f) * (1.f / 3.f);
            gx1 = (4.f * xn + 1.5f) * (1.f / 3.f);
        }
        float d0 = (fy0-gy0)*(fy0-gy0) + (fx0-gx0)*(fx0-gx0)
                 + (r0-gr)*(r0-gr) + (g0-gg)*(g0-gg) + (b0-gb)*(b0-gb);
        float d1 = (fy1-gy1)*(fy1-gy1) + (fx1-gx1)*(fx1-gx1);
        size_t o = (((size_t)b * K2 + k) * HL + yl) * WL + xl;
        K0[o] = __expf(-0.5f * d0);
        K1[o] = __expf(-0.5f * d1);
    }
}

// ---------------------------------------------------------------------------
// 3) logQ init: softmax(x) over C, channel-first (B,C,H,W)
__global__ void k_init_softmax(const float* __restrict__ xin, float* __restrict__ logQ) {
    int idx = blockIdx.x * blockDim.x + threadIdx.x;
    if (idx >= BB * HWF) return;
    int p = idx & (HWF - 1);
    int b = idx >> 18;
    const float* src = xin + (size_t)b * NC * HWF + p;
    float v[NC], mx = -1e30f;
    #pragma unroll
    for (int c = 0; c < NC; ++c) { v[c] = src[(size_t)c * HWF]; mx = fmaxf(mx, v[c]); }
    float s = 0.f;
    #pragma unroll
    for (int c = 0; c < NC; ++c) { v[c] = __expf(v[c] - mx); s += v[c]; }
    float inv = 1.f / s;
    float* dst = logQ + (size_t)b * NC * HWF + p;
    #pragma unroll
    for (int c = 0; c < NC; ++c) dst[(size_t)c * HWF] = v[c] * inv;
}

// ---------------------------------------------------------------------------
// 4) per-step: Qlow = avgpool4(softmax(logQ)) -> channel-LAST (B,HL,WL,C)
__global__ void k_pool_softmax(const float* __restrict__ logQ, float* __restrict__ Qlow) {
    int idx = blockIdx.x * blockDim.x + threadIdx.x;
    if (idx >= BB * HWL) return;
    int xl = idx & (WL - 1);
    int yl = (idx >> 7) & (HL - 1);
    int b  = idx >> 14;
    float acc[NC];
    #pragma unroll
    for (int c = 0; c < NC; ++c) acc[c] = 0.f;
    const float* base = logQ + (size_t)b * NC * HWF;
    for (int s4 = 0; s4 < 16; ++s4) {
        int y = 4 * yl + (s4 >> 2), x = 4 * xl + (s4 & 3);
        const float* src = base + y * WW + x;
        float v[NC], mx = -1e30f;
        #pragma unroll
        for (int c = 0; c < NC; ++c) { v[c] = src[(size_t)c * HWF]; mx = fmaxf(mx, v[c]); }
        float s = 0.f;
        #pragma unroll
        for (int c = 0; c < NC; ++c) { v[c] = __expf(v[c] - mx); s += v[c]; }
        float inv = 1.f / s;
        #pragma unroll
        for (int c = 0; c < NC; ++c) acc[c] += v[c] * inv;
    }
    float* dst = Qlow + ((size_t)idx) * NC;
    #pragma unroll
    for (int c = 0; c < NC; ++c) dst[c] = acc[c] * (1.f / 16.f);
}

// ---------------------------------------------------------------------------
// 5) PAC message + compat GEMM via WMMA, with async-staged LDS halo tile.
//    Block = 8 waves; wave w owns low-res row (blockIdx.y*8 + w) of a 16-px
//    wide column tile. The 18x26x16ch halo of Qlow is staged to LDS once
//    (GLOBAL_LOAD_ASYNC_TO_LDS_B128, OOB slots zero-filled -> no bounds
//    checks in the 121-tap loop). Lanes 0-15 accumulate the K0 message,
//    lanes 16-31 the K1 message; two chained v_wmma_f32_16x16x32_f16 apply
//    pw0*W0 and pw1*W1. msgLow channel-last (B,HL,WL,C).
__global__ void __launch_bounds__(256)
k_message(const float* __restrict__ Qlow,
          const float* __restrict__ K0, const float* __restrict__ K1,
          const float* __restrict__ wc0, const float* __restrict__ wc1,
          float* __restrict__ msgLow) {
    __shared__ float tile[TILE_FLOATS];   // [quarter][row][col][4ch]

    int lane = threadIdx.x & 31;
    int wave = threadIdx.x >> 5;          // 0..7 -> row within block
    int xt   = blockIdx.x;                // 0..7 column tile
    int yb   = blockIdx.y;                // 0..15 row block
    int b    = blockIdx.z;
    int yl   = yb * 8 + wave;
    int px   = xt * 16 + (lane & 15);

    // ---- stage Qlow halo: rows yb*8-5 .. yb*8+12, cols xt*16-5 .. xt*16+20
    {
        int y0 = yb * 8 - PADR;
        int x0 = xt * 16 - PADR;
        const float* qb = Qlow + (size_t)b * HWL * NC;
        for (int t = threadIdx.x; t < TPIX * 4; t += 256) {
            int q = t / TPIX;
            int r = (t % TPIX) / TCOLS;
            int c = t % TCOLS;
            float* dst = &tile[((q * TROWS + r) * TCOLS + c) * 4];
            int yn = y0 + r, xn = x0 + c;
            if (yn >= 0 && yn < HL && xn >= 0 && xn < WL) {
                const float* src = qb + ((size_t)yn * WL + xn) * NC + q * 4;
#if defined(HAVE_ASYNC_LDS)
                __builtin_amdgcn_global_load_async_to_lds_b128(
                    (g_b128_ptr)src, (l_b128_ptr)dst, 0, 0);
#else
                *(float4*)dst = *(const float4*)src;
#endif
            } else {
                float4 z; z.x = z.y = z.z = z.w = 0.f;
                *(float4*)dst = z;
            }
        }
    }
    wait_asynccnt0();
    __syncthreads();

    // ---- 121-tap PAC message from LDS (no bounds checks: halo is zeroed)
    const float* Kb = (lane < 16) ? K0 : K1;
    float pw        = (lane < 16) ? PW0 : PW1;

    float m[NC];
    #pragma unroll
    for (int c = 0; c < NC; ++c) m[c] = 0.f;

    int rr0 = wave + PADR;           // local row of this wave's pixel
    int cc0 = (lane & 15) + PADR;    // local col of this lane's pixel
    const size_t kpix = (size_t)yl * WL + px;
    for (int k = 0; k < K2; ++k) {
        int dy = k / KS - PADR, dx = k % KS - PADR;
        float kw = Kb[((size_t)b * K2 + k) * HWL + kpix];
        int po = (rr0 + dy) * TCOLS + (cc0 + dx);
        float4 q0 = *(const float4*)&tile[((0 * TROWS) * TCOLS * 4) + po * 4];
        float4 q1 = *(const float4*)&tile[((1 * TROWS) * TCOLS * 4) + po * 4];
        float4 q2 = *(const float4*)&tile[((2 * TROWS) * TCOLS * 4) + po * 4];
        float4 q3 = *(const float4*)&tile[((3 * TROWS) * TCOLS * 4) + po * 4];
        m[0]  += kw * q0.x; m[1]  += kw * q0.y; m[2]  += kw * q0.z; m[3]  += kw * q0.w;
        m[4]  += kw * q1.x; m[5]  += kw * q1.y; m[6]  += kw * q1.z; m[7]  += kw * q1.w;
        m[8]  += kw * q2.x; m[9]  += kw * q2.y; m[10] += kw * q2.z; m[11] += kw * q2.w;
        m[12] += kw * q3.x; m[13] += kw * q3.y; m[14] += kw * q3.z; m[15] += kw * q3.w;
    }
    #pragma unroll
    for (int c = 0; c < NC; ++c) m[c] *= pw;

    // ---- B-matrix (32x16 f16): lane n<16 holds column n, K=0..15 (classes);
    //      lanes 16-31 hold K=16..31 = zero padding.
    v16h B0, B1;
    bool lo = lane < 16;
    #pragma unroll
    for (int c = 0; c < NC; ++c) {
        B0[c] = lo ? (_Float16)m[c] : (_Float16)0.f;
        float v1 = __shfl(m[c], (lane & 15) + 16, 32);   // pull K1 message
        B1[c] = lo ? (_Float16)v1 : (_Float16)0.f;
    }
    // ---- A-matrix (16x32 f16): lane L<16 -> row L, K=0..15 in elems 0..15;
    //      lanes 16-31 hold K=16..31 = zero padding.
    int row = lane & 15;
    v16h A0, A1;
    #pragma unroll
    for (int i = 0; i < 16; ++i) {
        A0[i] = lo ? (_Float16)wc0[row * 16 + i] : (_Float16)0.f;
        A1[i] = lo ? (_Float16)wc1[row * 16 + i] : (_Float16)0.f;
    }

    v8f acc = {};
    acc = __builtin_amdgcn_wmma_f32_16x16x32_f16(false, A0, false, B0,
                                                 (short)0, acc, false, false);
    acc = __builtin_amdgcn_wmma_f32_16x16x32_f16(false, A1, false, B1,
                                                 (short)0, acc, false, false);

    // D layout: VGPR r, lanes 0-15 -> (class r, pixel lane),
    //           lanes 16-31 -> (class r+8, pixel lane-16)
    int coff = lo ? 0 : 8;
    float* dst = msgLow + (((size_t)b * HL + yl) * WL + px) * NC + coff;
    #pragma unroll
    for (int r = 0; r < 8; ++r) dst[r] = acc[r];
}

// ---------------------------------------------------------------------------
// 6) logQ = 0.8*softmax(x) - bilinear_upsample4(msgLow)   (align_corners=False)
__global__ void k_upsample(const float* __restrict__ xin,
                           const float* __restrict__ msgLow,
                           float* __restrict__ out) {
    int idx = blockIdx.x * blockDim.x + threadIdx.x;
    if (idx >= BB * HWF) return;
    int x = idx & (WW - 1);
    int y = (idx >> 9) & (HH - 1);
    int b = idx >> 18;

    const float* src = xin + (size_t)b * NC * HWF + y * WW + x;
    float u[NC], mx = -1e30f;
    #pragma unroll
    for (int c = 0; c < NC; ++c) { u[c] = src[(size_t)c * HWF]; mx = fmaxf(mx, u[c]); }
    float s = 0.f;
    #pragma unroll
    for (int c = 0; c < NC; ++c) { u[c] = __expf(u[c] - mx); s += u[c]; }
    float inv = 1.f / s;

    float sy = y * 0.25f - 0.375f;
    float sx = x * 0.25f - 0.375f;
    int y0 = (int)floorf(sy), x0 = (int)floorf(sx);
    float wy = sy - y0, wx = sx - x0;
    int y0c = min(max(y0, 0), HL - 1), y1c = min(max(y0 + 1, 0), HL - 1);
    int x0c = min(max(x0, 0), WL - 1), x1c = min(max(x0 + 1, 0), WL - 1);
    float w00 = (1.f - wy) * (1.f - wx), w01 = (1.f - wy) * wx;
    float w10 = wy * (1.f - wx),        w11 = wy * wx;

    const float* mb  = msgLow + (size_t)b * HWL * NC;
    const float* p00 = mb + ((size_t)y0c * WL + x0c) * NC;
    const float* p01 = mb + ((size_t)y0c * WL + x1c) * NC;
    const float* p10 = mb + ((size_t)y1c * WL + x0c) * NC;
    const float* p11 = mb + ((size_t)y1c * WL + x1c) * NC;

    float* dst = out + (size_t)b * NC * HWF + y * WW + x;
    #pragma unroll
    for (int c = 0; c < NC; ++c) {
        float msg = w00 * p00[c] + w01 * p01[c] + w10 * p10[c] + w11 * p11[c];
        dst[(size_t)c * HWF] = UNARY_W * (u[c] * inv) - msg;
    }
}

// ---------------------------------------------------------------------------
extern "C" void kernel_launch(void* const* d_in, const int* in_sizes, int n_in,
                              void* d_out, int out_size, void* d_ws, size_t ws_size,
                              hipStream_t stream) {
    const float* x    = (const float*)d_in[0];   // (2,16,512,512)
    const float* img  = (const float*)d_in[1];   // (2,3,512,512)
    const float* wc0  = (const float*)d_in[2];   // (16,16)
    const float* wc1  = (const float*)d_in[3];   // (16,16)
    float* out        = (float*)d_out;           // (2,16,512,512)

    float* ws = (float*)d_ws;
    size_t off = 0;
    float* logQ   = ws + off; off += (size_t)BB * NC * HWF;   // 8388608
    float* K0     = ws + off; off += (size_t)BB * K2 * HWL;   // 3964928
    float* K1     = ws + off; off += (size_t)BB * K2 * HWL;   // 3964928
    float* Qlow   = ws + off; off += (size_t)BB * HWL * NC;   //  524288
    float* msgLow = ws + off; off += (size_t)BB * HWL * NC;   //  524288
    float* imgLow = ws + off; off += (size_t)BB * 3 * HWL;    //   98304

    {
        int n = BB * 3 * HWL;
        k_pool_image<<<(n + 255) / 256, 256, 0, stream>>>(img, imgLow);
    }
    {
        int n = BB * HWL;
        k_gauss<<<(n + 127) / 128, 128, 0, stream>>>(imgLow, K0, K1);
    }
    {
        int n = BB * HWF;
        k_init_softmax<<<(n + 127) / 128, 128, 0, stream>>>(x, logQ);
    }

    for (int step = 0; step < 5; ++step) {
        {
            int n = BB * HWL;
            k_pool_softmax<<<(n + 127) / 128, 128, 0, stream>>>(logQ, Qlow);
        }
        {
            dim3 grid(WL / 16, HL / 8, BB);   // 8 x 16 x 2 blocks, 256 thr
            k_message<<<grid, 256, 0, stream>>>(Qlow, K0, K1, wc0, wc1, msgLow);
        }
        {
            int n = BB * HWF;
            float* dst = (step == 4) ? out : logQ;
            k_upsample<<<(n + 127) / 128, 128, 0, stream>>>(x, msgLow, dst);
        }
    }
}
